// AlignmentAttention_62654982914793
// MI455X (gfx1250) — compile-verified, matrix operand order checked by reading, not run on hardware
//
#include <hip/hip_runtime.h>
#include <hip/hip_bf16.h>

#define BATCH 4
#define HEADS 8
#define TT 256
#define SSZ 256
#define DK 64
#define BH (BATCH*HEADS)
#define NEGV (-100000000.0f)

typedef __attribute__((ext_vector_type(16))) _Float16 v16h;
typedef __attribute__((ext_vector_type(8)))  float    v8f;

static __device__ inline v8f wmma_f16(v16h a, v16h b, v8f c) {
    // D = A(16x32 f16) * B(32x16 f16) + C(16x16 f32)
    return __builtin_amdgcn_wmma_f32_16x16x32_f16(false, a, false, b, (short)0, c, false, false);
}

static __device__ inline v8f zero8() { v8f z = {}; return z; }

// K-pair base for 16-bit A/B operand VGPR v (0..7), half h (0/1).
// V0: lo K=0,1 / hi K=8,9 ... V4: lo K=16,17 / hi K=24,25 ...
static __device__ inline int kb_of(int v, int h) {
    return (v < 4) ? (2 * v + 8 * h) : (16 + 2 * (v - 4) + 8 * h);
}

// Operand element (row = lane&15, K = k0+kb) = src[(lane&15)*ld + k0 + kb]  (f32 -> f16)
static __device__ inline v16h load_op_rowmajor_f32(const float* __restrict__ src, int ld, int lane, int k0) {
    int lo = lane & 15, h = lane >> 4;
    const float* p = src + lo * ld + k0;
    v16h r;
#pragma unroll
    for (int v = 0; v < 8; ++v) {
        int kb = kb_of(v, h);
        r[2 * v]     = (_Float16)p[kb];
        r[2 * v + 1] = (_Float16)p[kb + 1];
    }
    return r;
}

// Operand element (K = k0+kb, col = lane&15) = src[(k0+kb)*ld + (lane&15)]  (f32 -> f16)
static __device__ inline v16h load_op_colmajor_f32(const float* __restrict__ src, int ld, int lane, int k0) {
    int lo = lane & 15, h = lane >> 4;
    const float* p = src + lo;
    v16h r;
#pragma unroll
    for (int v = 0; v < 8; ++v) {
        int kb = k0 + kb_of(v, h);
        r[2 * v]     = (_Float16)p[kb * ld];
        r[2 * v + 1] = (_Float16)p[(kb + 1) * ld];
    }
    return r;
}

// A operand from f16 LDS, row-major with leading dim ld
static __device__ inline v16h load_op_lds_f16(const _Float16* src, int ld, int lane, int k0) {
    int lo = lane & 15, h = lane >> 4;
    const _Float16* p = src + lo * ld + k0;
    v16h r;
#pragma unroll
    for (int v = 0; v < 8; ++v) {
        int kb = kb_of(v, h);
        r[2 * v]     = p[kb];
        r[2 * v + 1] = p[kb + 1];
    }
    return r;
}

// Reductions over 16-lane halves (xor masks <= 8 keep the halves separate,
// matching the C/D layout where each half-wave owns distinct rows).
static __device__ inline float rmax16(float x) {
    x = fmaxf(x, __shfl_xor(x, 1, 32));
    x = fmaxf(x, __shfl_xor(x, 2, 32));
    x = fmaxf(x, __shfl_xor(x, 4, 32));
    x = fmaxf(x, __shfl_xor(x, 8, 32));
    return x;
}
static __device__ inline float rsum16(float x) {
    x += __shfl_xor(x, 1, 32);
    x += __shfl_xor(x, 2, 32);
    x += __shfl_xor(x, 4, 32);
    x += __shfl_xor(x, 8, 32);
    return x;
}

static __device__ inline float fast_tanh(float x) {
#if __has_builtin(__builtin_amdgcn_tanhf)
    return __builtin_amdgcn_tanhf(x);   // v_tanh_f32 (gfx1250 TRANS op)
#else
    return tanhf(x);
#endif
}

static __device__ inline float fast_sigmoid(float x) {
    return 1.0f / (1.0f + __expf(-x));
}

// ---------------------------------------------------------------------------
// Kernel 1: qp = Q @ Wq^T, kp = K @ Wk^T (WMMA), gate = sigmoid(Q @ Wg^T + bg)
// grid = BH blocks, 256 threads (8 waves); each wave does 4 row-tiles.
// ---------------------------------------------------------------------------
__global__ __launch_bounds__(256) void proj_kernel(
    const float* __restrict__ Q, const float* __restrict__ K,
    const float* __restrict__ Wq, const float* __restrict__ Wk,
    const float* __restrict__ Wg, const float* __restrict__ bg,
    float* __restrict__ qp, float* __restrict__ kp, float* __restrict__ gate)
{
    int bh = blockIdx.x;
    int wave = threadIdx.x >> 5;
    int lane = threadIdx.x & 31;
    int lo = lane & 15, h = lane >> 4;

#pragma unroll
    for (int i = 0; i < 4; ++i) {
        int tile = wave + 8 * i;            // 0..15: Q->qp, 16..31: K->kp
        bool isK = tile >= 16;
        int t0 = (tile & 15) * 16;
        const float* src = isK ? K : Q;
        const float* W   = isK ? Wk : Wq;
        float* dst       = isK ? kp : qp;

        const float* Sb = src + (bh * 256 + t0) * DK;
        v16h a0 = load_op_rowmajor_f32(Sb, DK, lane, 0);
        v16h a1 = load_op_rowmajor_f32(Sb, DK, lane, 32);
#pragma unroll
        for (int et = 0; et < 4; ++et) {
            // B(k=d, n=e) = W[e*DK + d]  (einsum 'ed' with d contracted)
            v8f acc = zero8();
            acc = wmma_f16(a0, load_op_rowmajor_f32(W + 16 * et * DK, DK, lane, 0),  acc);
            acc = wmma_f16(a1, load_op_rowmajor_f32(W + 16 * et * DK, DK, lane, 32), acc);
#pragma unroll
            for (int r = 0; r < 8; ++r) {
                int t = t0 + r + 8 * h;
                dst[(bh * 256 + t) * DK + 16 * et + lo] = acc[r];
            }
        }
    }
    // gate: one thread per row
    int t = threadIdx.x;
    const float* qrow = Q + (bh * TT + t) * DK;
    float g = bg[0];
#pragma unroll 8
    for (int d = 0; d < DK; ++d) g += qrow[d] * Wg[d];
    gate[bh * TT + t] = fast_sigmoid(g);
}

// ---------------------------------------------------------------------------
// Kernel 2: dot branch. One wave per (bh, 16-row t-tile).
// scores (16x256) live in 128 VGPRs; P staged in LDS as f16 for the A operand
// of P@V.
// ---------------------------------------------------------------------------
__global__ __launch_bounds__(32) void dot_attn_kernel(
    const float* __restrict__ Q, const float* __restrict__ K,
    const float* __restrict__ V, const int* __restrict__ mask,
    float* __restrict__ dot_score, float* __restrict__ dot_out)
{
    __shared__ _Float16 Plds[16][256];     // 8 KB
    int t0 = blockIdx.x * 16;
    int bh = blockIdx.y;
    int bidx = bh >> 3;                    // batch index for mask (B,1,T,S)
    int lane = threadIdx.x;
    int lo = lane & 15, h = lane >> 4;

    const float* Qb = Q + (bh * TT + t0) * DK;
    v16h a0 = load_op_rowmajor_f32(Qb, DK, lane, 0);
    v16h a1 = load_op_rowmajor_f32(Qb, DK, lane, 32);

    v8f sc[16];
#pragma unroll
    for (int j = 0; j < 16; ++j) {
        const float* Kb = K + (bh * SSZ + j * 16) * DK;   // B(k=d,n=s)=K[s*DK+d]
        v8f acc = zero8();
        acc = wmma_f16(a0, load_op_rowmajor_f32(Kb, DK, lane, 0),  acc);
        acc = wmma_f16(a1, load_op_rowmajor_f32(Kb, DK, lane, 32), acc);
#pragma unroll
        for (int r = 0; r < 8; ++r) {
            int t = t0 + r + 8 * h;
            int s = j * 16 + lo;
            int m = mask[(bidx * TT + t) * SSZ + s];
            sc[j][r] = (m == 0) ? NEGV : acc[r] * 0.125f;   // 1/sqrt(64)
        }
    }

    // masked softmax over s (rows live across the 16-lane half)
    float mx[8], sm[8];
#pragma unroll
    for (int r = 0; r < 8; ++r) {
        float pm = sc[0][r];
#pragma unroll
        for (int j = 1; j < 16; ++j) pm = fmaxf(pm, sc[j][r]);
        mx[r] = rmax16(pm);
    }
#pragma unroll
    for (int r = 0; r < 8; ++r) {
        float ps = 0.0f;
#pragma unroll
        for (int j = 0; j < 16; ++j) {
            float e = __expf(sc[j][r] - mx[r]);
            sc[j][r] = e;
            ps += e;
        }
        sm[r] = 1.0f / rsum16(ps);
    }

    // normalized probs -> d_out (dot_score) and LDS (f16 A operand for P@V)
#pragma unroll
    for (int j = 0; j < 16; ++j)
#pragma unroll
        for (int r = 0; r < 8; ++r) {
            float p = sc[j][r] * sm[r];
            int t = t0 + r + 8 * h, s = j * 16 + lo;
            dot_score[(bh * TT + t) * SSZ + s] = p;
            Plds[r + 8 * h][s] = (_Float16)p;
        }
    __syncthreads();

    // out = P @ V  (A = P from LDS, B(k=s,n=d) = V[s*DK+d] col-major loader)
    v8f o0 = zero8(), o1 = zero8(), o2 = zero8(), o3 = zero8();
#pragma unroll
    for (int kk = 0; kk < 8; ++kk) {
        v16h aP = load_op_lds_f16(&Plds[0][0], 256, lane, kk * 32);
        const float* Vb = V + bh * SSZ * DK;
        o0 = wmma_f16(aP, load_op_colmajor_f32(Vb + 0,  DK, lane, kk * 32), o0);
        o1 = wmma_f16(aP, load_op_colmajor_f32(Vb + 16, DK, lane, kk * 32), o1);
        o2 = wmma_f16(aP, load_op_colmajor_f32(Vb + 32, DK, lane, kk * 32), o2);
        o3 = wmma_f16(aP, load_op_colmajor_f32(Vb + 48, DK, lane, kk * 32), o3);
    }
#pragma unroll
    for (int r = 0; r < 8; ++r) {
        int t = t0 + r + 8 * h;
        float* dst = dot_out + (bh * TT + t) * DK + lo;
        dst[0]  = o0[r];
        dst[16] = o1[r];
        dst[32] = o2[r];
        dst[48] = o3[r];
    }
}

// ---------------------------------------------------------------------------
// Kernel 3: additive branch + gated combine. One wave per (bh, 16-row t-tile).
// add_sc[t,s] = sum_e v[e]*tanh(qp[t,e]+kp[s,e])  (TRANS-unit bound)
// ---------------------------------------------------------------------------
__global__ __launch_bounds__(32) void add_attn_kernel(
    const float* __restrict__ V, const int* __restrict__ mask,
    const float* __restrict__ vw,
    const float* __restrict__ qp, const float* __restrict__ kp,
    const float* __restrict__ gate, const float* __restrict__ dot_out,
    float* __restrict__ out)
{
    __shared__ float Qp[16][65];           // padded: conflict-free row reads
    __shared__ float Kp[16][65];
    __shared__ float Vw[64];
    __shared__ _Float16 Plds[16][256];

    int t0 = blockIdx.x * 16;
    int bh = blockIdx.y;
    int bidx = bh >> 3;
    int lane = threadIdx.x;
    int lo = lane & 15, h = lane >> 4;

    for (int i = lane; i < 1024; i += 32)
        Qp[i >> 6][i & 63] = qp[(bh * TT + t0) * DK + i];
    if (lane < 32) { Vw[lane] = vw[lane]; Vw[lane + 32] = vw[lane + 32]; }
    __syncthreads();

    v8f sc[16];
    for (int j = 0; j < 16; ++j) {
        __syncthreads();
        for (int i = lane; i < 1024; i += 32)
            Kp[i >> 6][i & 63] = kp[(bh * SSZ + j * 16) * DK + i];
        __syncthreads();

        const float* krow = &Kp[lo][0];
#pragma unroll
        for (int r = 0; r < 8; ++r) {
            const float* qrow = &Qp[r + 8 * h][0];
            float acc = 0.0f;
#pragma unroll 8
            for (int e = 0; e < 64; ++e)
                acc += Vw[e] * fast_tanh(qrow[e] + krow[e]);
            int t = t0 + r + 8 * h, s = j * 16 + lo;
            int m = mask[(bidx * TT + t) * SSZ + s];
            sc[j][r] = (m == 0) ? NEGV : acc;
        }
    }

    // masked softmax
    float mx[8], sm[8];
#pragma unroll
    for (int r = 0; r < 8; ++r) {
        float pm = sc[0][r];
#pragma unroll
        for (int j = 1; j < 16; ++j) pm = fmaxf(pm, sc[j][r]);
        mx[r] = rmax16(pm);
    }
#pragma unroll
    for (int r = 0; r < 8; ++r) {
        float ps = 0.0f;
#pragma unroll
        for (int j = 0; j < 16; ++j) {
            float e = __expf(sc[j][r] - mx[r]);
            sc[j][r] = e;
            ps += e;
        }
        sm[r] = 1.0f / rsum16(ps);
    }
#pragma unroll
    for (int j = 0; j < 16; ++j)
#pragma unroll
        for (int r = 0; r < 8; ++r)
            Plds[r + 8 * h][j * 16 + lo] = (_Float16)(sc[j][r] * sm[r]);
    __syncthreads();

    // add_out = P @ V via WMMA
    v8f o0 = zero8(), o1 = zero8(), o2 = zero8(), o3 = zero8();
#pragma unroll
    for (int kk = 0; kk < 8; ++kk) {
        v16h aP = load_op_lds_f16(&Plds[0][0], 256, lane, kk * 32);
        const float* Vb = V + bh * SSZ * DK;
        o0 = wmma_f16(aP, load_op_colmajor_f32(Vb + 0,  DK, lane, kk * 32), o0);
        o1 = wmma_f16(aP, load_op_colmajor_f32(Vb + 16, DK, lane, kk * 32), o1);
        o2 = wmma_f16(aP, load_op_colmajor_f32(Vb + 32, DK, lane, kk * 32), o2);
        o3 = wmma_f16(aP, load_op_colmajor_f32(Vb + 48, DK, lane, kk * 32), o3);
    }

    // gated combine: out = g*dot_out + (1-g)*add_out
#pragma unroll
    for (int r = 0; r < 8; ++r) {
        int t = t0 + r + 8 * h;
        float g = gate[bh * TT + t];
        const float* dsrc = dot_out + (bh * TT + t) * DK + lo;
        float* dst = out + (bh * TT + t) * DK + lo;
        dst[0]  = g * dsrc[0]  + (1.0f - g) * o0[r];
        dst[16] = g * dsrc[16] + (1.0f - g) * o1[r];
        dst[32] = g * dsrc[32] + (1.0f - g) * o2[r];
        dst[48] = g * dsrc[48] + (1.0f - g) * o3[r];
    }
}

extern "C" void kernel_launch(void* const* d_in, const int* in_sizes, int n_in,
                              void* d_out, int out_size, void* d_ws, size_t ws_size,
                              hipStream_t stream) {
    (void)in_sizes; (void)n_in; (void)out_size; (void)ws_size;
    const float* Q    = (const float*)d_in[0];
    const float* K    = (const float*)d_in[1];
    const float* V    = (const float*)d_in[2];
    const int*   mask = (const int*)d_in[3];
    const float* Wq   = (const float*)d_in[4];
    const float* Wk   = (const float*)d_in[5];
    const float* vw   = (const float*)d_in[6];
    const float* Wg   = (const float*)d_in[7];
    const float* bg   = (const float*)d_in[8];

    float* out       = (float*)d_out;                 // [B,H,T,DK]
    float* dot_score = out + (size_t)BH * TT * DK;    // [B,H,T,S]

    float* qp      = (float*)d_ws;                    // [BH,T,DK]
    float* kp      = qp + (size_t)BH * TT * DK;       // [BH,S,DK]
    float* gate    = kp + (size_t)BH * SSZ * DK;      // [BH,T]
    float* dot_out = gate + (size_t)BH * TT;          // [BH,T,DK]

    proj_kernel<<<BH, 256, 0, stream>>>(Q, K, Wq, Wk, Wg, bg, qp, kp, gate);
    dot_attn_kernel<<<dim3(TT / 16, BH), 32, 0, stream>>>(Q, K, V, mask, dot_score, dot_out);
    add_attn_kernel<<<dim3(TT / 16, BH), 32, 0, stream>>>(V, mask, vw, qp, kp, gate, dot_out, out);
}
